// MSWIDT_6828998001252
// MI455X (gfx1250) — matrix operand order repaired and checked by reading, not compiled, not run on hardware
//
#include <hip/hip_runtime.h>
#include <hip/hip_bf16.h>
#include <math.h>

typedef _Float16 half_t;
typedef __attribute__((ext_vector_type(16))) _Float16 v16h;
typedef __attribute__((ext_vector_type(8)))  _Float16 v8h;
typedef __attribute__((ext_vector_type(8)))  float    v8f;
typedef unsigned int u32;
typedef __attribute__((ext_vector_type(4))) unsigned int u32x4;
typedef __attribute__((ext_vector_type(8))) int i32x8;
typedef __attribute__((ext_vector_type(4))) int i32x4;

#define T_TOK  32768
#define DIM    512
#define NH     8
#define HD     64
#define WIN    128
#define NWIN   256
#define PHASES 6
#define VOCAB  256
#define DFF    2048
#define BATCH  8
#define SEQ    4096

__device__ __forceinline__ float gelu_f(float x) {
    float x3 = x * x * x;
    return 0.5f * x * (1.0f + tanhf(0.7978845608028654f * (x + 0.044715f * x3)));
}

__device__ __forceinline__ v16h mk16(v8h lo, v8h hi) {
    union { v16h v; v8h p[2]; } u;
    u.p[0] = lo; u.p[1] = hi;
    return u.v;
}

__device__ __forceinline__ v8f zero8() {
    v8f z = {0.f, 0.f, 0.f, 0.f, 0.f, 0.f, 0.f, 0.f};
    return z;
}

// ---------------------------------------------------------------- TDM
// DMA one 2D f16 tile (32 K-elems x 128 rows, row stride = K elems) into LDS.
// D# packing per CDNA5 ISA 8.3/8.4. Groups 2/3 zero (<=2D tensor).
__device__ __forceinline__ void tdm_load_tile(u32 ldsAddr, const half_t* gptr, int K) {
    unsigned long long ga = (unsigned long long)(size_t)gptr;
    u32x4 g0;
    g0[0] = 1u;                                   // count=1, user mode
    g0[1] = ldsAddr;                              // lds_addr (bytes)
    g0[2] = (u32)(ga & 0xFFFFFFFFu);              // global_addr[31:0]
    g0[3] = (u32)((ga >> 32) & 0x01FFFFFFu)       // global_addr[56:32]
          | (2u << 30);                           // type = 2 ("image")
    i32x8 g1 = {0, 0, 0, 0, 0, 0, 0, 0};
    g1[0] = (int)(1u << 16);                      // data_size = 1 -> 2 bytes
    g1[1] = (int)(((u32)K & 0xFFFFu) << 16);      // tensor_dim0[15:0] @ bits 63:48
    g1[2] = (int)((((u32)K >> 16) & 0xFFFFu)      // tensor_dim0[31:16]
          | (128u << 16));                        // tensor_dim1[15:0] = 128
    g1[3] = (int)(32u << 16);                     // tile_dim0 = 32 @ bits 127:112
    g1[4] = (int)128u;                            // tile_dim1 = 128
    g1[5] = (int)(u32)K;                          // tensor_dim0_stride[31:0] = K
    i32x4 gz = {0, 0, 0, 0};
#if defined(__clang_major__) && (__clang_major__ >= 23)
    i32x8 gz8 = {0, 0, 0, 0, 0, 0, 0, 0};
    __builtin_amdgcn_tensor_load_to_lds(g0, g1, gz, gz, gz8, 0);
#else
    __builtin_amdgcn_tensor_load_to_lds(g0, g1, gz, gz, 0);
#endif
}

// ---------------------------------------------------------------- weights
// src: (rows x cols) f32 row-major (in x out). dst: (cols x rows) f16 (out x in).
__global__ void pack_weight_kernel(const float* __restrict__ src, half_t* __restrict__ dst,
                                   int rows, int cols) {
    int idx = blockIdx.x * blockDim.x + threadIdx.x;
    if (idx >= rows * cols) return;
    int n = idx / rows;     // output column
    int k = idx % rows;     // input row
    dst[idx] = (half_t)src[k * cols + n];
}

// t = gelu(mask_ratio_t @ t_W + t_b) : (B, D)
__global__ void tvec_kernel(const float* __restrict__ mr, const float* __restrict__ tW,
                            const float* __restrict__ tb, float* __restrict__ tvec) {
    int idx = blockIdx.x * blockDim.x + threadIdx.x;
    if (idx >= BATCH * DIM) return;
    int b = idx / DIM, d = idx % DIM;
    tvec[idx] = gelu_f(mr[b] * tW[d] + tb[d]);
}

// ---------------------------------------------------------------- embedding
__global__ void embed_kernel(const int* __restrict__ x, const float* __restrict__ tok,
                             const float* __restrict__ pos, const float* __restrict__ maskemb,
                             const float* __restrict__ tvec, float* __restrict__ h) {
    int idx = blockIdx.x * blockDim.x + threadIdx.x;   // T_TOK*DIM grid, exact
    int t = idx >> 9;          // / DIM
    int d = idx & (DIM - 1);
    int b = t / SEQ;
    int n = t - b * SEQ;
    int tokid = x[t];
    int m = (tokid == 1) ? 1 : 0;
    h[idx] = tok[tokid * DIM + d] + pos[n * DIM + d] + maskemb[m * DIM + d] + tvec[b * DIM + d];
}

__global__ void phase_begin_kernel(float* __restrict__ h, float* __restrict__ pin,
                                   const float* __restrict__ pe) {
    int idx = blockIdx.x * blockDim.x + threadIdx.x;
    int d = idx & (DIM - 1);
    float v = h[idx] + pe[d];
    h[idx] = v;
    pin[idx] = v;
}

__global__ void phase_end_kernel(float* __restrict__ h, const float* __restrict__ pin) {
    int idx = blockIdx.x * blockDim.x + threadIdx.x;
    h[idx] = 0.5f * (pin[idx] + h[idx]);
}

__global__ void tof16_kernel(const float* __restrict__ h, half_t* __restrict__ o) {
    int idx = blockIdx.x * blockDim.x + threadIdx.x;
    o[idx] = (half_t)h[idx];
}

// ---------------------------------------------------------------- layernorm
// one wave per token; lane holds 16 contiguous f32
__global__ __launch_bounds__(256) void ln_kernel(const float* __restrict__ h,
                                                 const float* __restrict__ s,
                                                 const float* __restrict__ b,
                                                 half_t* __restrict__ out) {
    const int wv = threadIdx.x >> 5;
    const int lane = threadIdx.x & 31;
    const int t = blockIdx.x * 8 + wv;
    const float* row = h + (size_t)t * DIM;
    float x[16];
    const float4* rp = (const float4*)row + lane * 4;
    float sum = 0.f, sq = 0.f;
#pragma unroll
    for (int i = 0; i < 4; i++) {
        float4 v = rp[i];
        x[i * 4 + 0] = v.x; x[i * 4 + 1] = v.y; x[i * 4 + 2] = v.z; x[i * 4 + 3] = v.w;
    }
#pragma unroll
    for (int i = 0; i < 16; i++) { sum += x[i]; sq += x[i] * x[i]; }
#pragma unroll
    for (int off = 1; off < 32; off <<= 1) {
        sum += __shfl_xor(sum, off);
        sq  += __shfl_xor(sq, off);
    }
    float mean = sum * (1.f / 512.f);
    float var  = sq * (1.f / 512.f) - mean * mean;
    float rstd = rsqrtf(var + 1e-6f);
    half_t* op = out + (size_t)t * DIM + lane * 16;
#pragma unroll
    for (int i = 0; i < 16; i++) {
        int d = lane * 16 + i;
        op[i] = (half_t)((x[i] - mean) * rstd * s[d] + b[d]);
    }
}

// ---------------------------------------------------------------- WMMA GEMM
// C(MxN) = A(MxK, f16) * BT(NxK, f16)^T + bias
// Tiles staged by TDM (tensor_load_to_lds) with double-buffered LDS.
// EPI: 0 = f16 store, 1 = f16 gelu store, 2 = f32 in-place residual add, 3 = f32 store
template <int EPI>
__global__ __launch_bounds__(256) void gemm_kernel(const half_t* __restrict__ A,
                                                   const half_t* __restrict__ BT,
                                                   const float* __restrict__ bias,
                                                   void* __restrict__ outv,
                                                   int K, int N) {
    __shared__ __align__(16) half_t sA[2][128 * 32];
    __shared__ __align__(16) half_t sB[2][128 * 32];
    const int tid  = threadIdx.x;
    const int lane = tid & 31;
    const int wv   = tid >> 5;
    const int mBase = blockIdx.x * 128;
    const int nBase = blockIdx.y * 128;
    const int wm = (wv >> 1) * 32;     // wave M offset within tile
    const int wn = (wv & 1) * 64;      // wave N offset within tile
    const int l15  = lane & 15;
    const int aoff = (lane < 16) ? 0 : 8;
    const int boff = (lane < 16) ? 0 : 8;

    const u32 aAddr[2] = { (u32)(size_t)&sA[0][0], (u32)(size_t)&sA[1][0] };
    const u32 bAddr[2] = { (u32)(size_t)&sB[0][0], (u32)(size_t)&sB[1][0] };
    const half_t* aTile = A + (size_t)mBase * K;
    const half_t* bTile = BT + (size_t)nBase * K;

    v8f acc[2][4];
#pragma unroll
    for (int mi = 0; mi < 2; mi++)
#pragma unroll
        for (int ni = 0; ni < 4; ni++) acc[mi][ni] = zero8();

    const int nk = K >> 5;
    if (wv == 0) {                      // wave-uniform: exactly one DMA per tile
        tdm_load_tile(aAddr[0], aTile, K);
        tdm_load_tile(bAddr[0], bTile, K);
    }

    for (int ks = 0; ks < nk; ks++) {
        if (wv == 0) __builtin_amdgcn_s_wait_tensorcnt(0);  // current tile landed
        __syncthreads();                                    // publish tile to all waves
        if ((ks + 1) < nk && wv == 0) {                     // prefetch next tile
            tdm_load_tile(aAddr[(ks + 1) & 1], aTile + (size_t)(ks + 1) * 32, K);
            tdm_load_tile(bAddr[(ks + 1) & 1], bTile + (size_t)(ks + 1) * 32, K);
        }
        const half_t* cA = sA[ks & 1];
        const half_t* cB = sB[ks & 1];

        v16h af[2];
#pragma unroll
        for (int mi = 0; mi < 2; mi++) {
            const half_t* p = cA + (wm + mi * 16 + l15) * 32;
            af[mi] = mk16(*(const v8h*)(p + aoff), *(const v8h*)(p + 16 + aoff));
        }
#pragma unroll
        for (int ni = 0; ni < 4; ni++) {
            const half_t* p = cB + (wn + ni * 16 + l15) * 32 + boff;
            v16h bf = mk16(*(const v8h*)p, *(const v8h*)(p + 16));
#pragma unroll
            for (int mi = 0; mi < 2; mi++)
                acc[mi][ni] = __builtin_amdgcn_wmma_f32_16x16x32_f16(
                    false, af[mi], false, bf, (short)0, acc[mi][ni], false, false);
        }
        __syncthreads();     // all reads of this buffer done before it is re-filled
    }

#pragma unroll
    for (int mi = 0; mi < 2; mi++)
#pragma unroll
        for (int ni = 0; ni < 4; ni++) {
            int col = nBase + wn + ni * 16 + l15;
            float bs = bias[col];
#pragma unroll
            for (int r = 0; r < 8; r++) {
                int row = mBase + wm + mi * 16 + r + ((lane < 16) ? 0 : 8);
                float val = acc[mi][ni][r] + bs;
                size_t o = (size_t)row * N + col;
                if constexpr (EPI == 0)      ((half_t*)outv)[o] = (half_t)val;
                else if constexpr (EPI == 1) ((half_t*)outv)[o] = (half_t)gelu_f(val);
                else if constexpr (EPI == 2) ((float*)outv)[o] += val;
                else                         ((float*)outv)[o] = val;
            }
        }
}

// ---------------------------------------------------------------- attention
// one block per (window, head). 8 waves; wave owns 16 query rows. flash-style.
// manual staging (gather wraps mod SEQ for shifted phases -> not a strided D# tile)
__global__ __launch_bounds__(256) void attn_kernel(const half_t* __restrict__ q,
                                                   const half_t* __restrict__ k,
                                                   const half_t* __restrict__ v,
                                                   half_t* __restrict__ ao, int shift) {
    __shared__ __align__(16) half_t sQ[WIN * HD];
    __shared__ __align__(16) half_t sK[WIN * HD];
    __shared__ __align__(16) half_t sVt[HD * WIN];
    __shared__ __align__(16) half_t sP[8 * 16 * 32];

    const int tid = threadIdx.x, lane = tid & 31, wv = tid >> 5;
    const int b  = blockIdx.x >> 5;
    const int w  = blockIdx.x & 31;
    const int hh = blockIdx.y;
    const int colBase = hh * HD;

    // stage q/k/v tiles: thread loads 32 halves of one token row
    const int li = tid >> 1;
    const int lc = (tid & 1) * 32;
    const int tokg = b * SEQ + ((w * WIN + li + shift) & (SEQ - 1));
    {
        const uint4* qg = (const uint4*)(q + (size_t)tokg * DIM + colBase + lc);
        const uint4* kg = (const uint4*)(k + (size_t)tokg * DIM + colBase + lc);
        const uint4* vg = (const uint4*)(v + (size_t)tokg * DIM + colBase + lc);
#pragma unroll
        for (int j = 0; j < 4; j++) {
            *(uint4*)(sQ + li * HD + lc + j * 8) = qg[j];
            *(uint4*)(sK + li * HD + lc + j * 8) = kg[j];
        }
        union { uint4 u[4]; half_t h[32]; } vb;
#pragma unroll
        for (int j = 0; j < 4; j++) vb.u[j] = vg[j];
#pragma unroll
        for (int c = 0; c < 32; c++) sVt[(lc + c) * WIN + li] = vb.h[c];
    }
    __syncthreads();

    const int qr = lane & 15;
    const bool lowhalf = (lane < 16);
    const int aoff = lowhalf ? 0 : 8;
    const int boff = lowhalf ? 0 : 8;

    float m[8], l[8];
    v8f O[4];
#pragma unroll
    for (int r = 0; r < 8; r++) { m[r] = -1e30f; l[r] = 0.f; }
#pragma unroll
    for (int t = 0; t < 4; t++) O[t] = zero8();

    v16h aq[2];
#pragma unroll
    for (int ks = 0; ks < 2; ks++) {
        const half_t* p = sQ + (wv * 16 + qr) * HD + ks * 32;
        aq[ks] = mk16(*(const v8h*)(p + aoff), *(const v8h*)(p + 16 + aoff));
    }

    for (int jp = 0; jp < 4; jp++) {           // pairs of 16-key tiles (32 keys)
        v8f s0 = zero8(), s1 = zero8();
#pragma unroll
        for (int ks = 0; ks < 2; ks++) {
            const half_t* pb0 = sK + (jp * 32 + qr) * HD + ks * 32 + boff;
            v16h bf0 = mk16(*(const v8h*)pb0, *(const v8h*)(pb0 + 16));
            const half_t* pb1 = sK + (jp * 32 + 16 + qr) * HD + ks * 32 + boff;
            v16h bf1 = mk16(*(const v8h*)pb1, *(const v8h*)(pb1 + 16));
            s0 = __builtin_amdgcn_wmma_f32_16x16x32_f16(false, aq[ks], false, bf0, (short)0, s0, false, false);
            s1 = __builtin_amdgcn_wmma_f32_16x16x32_f16(false, aq[ks], false, bf1, (short)0, s1, false, false);
        }
        // online softmax over this 32-key slab, stage P in A-fragment layout
        half_t* spw = sP + wv * 512;
#pragma unroll
        for (int r = 0; r < 8; r++) {
            float v0 = s0[r] * 0.125f, v1 = s1[r] * 0.125f;
            float tm = fmaxf(v0, v1);
#pragma unroll
            for (int off = 1; off < 16; off <<= 1) tm = fmaxf(tm, __shfl_xor(tm, off));
            float mn = fmaxf(m[r], tm);
            float p0 = __expf(v0 - mn), p1 = __expf(v1 - mn);
            float ts = p0 + p1;
#pragma unroll
            for (int off = 1; off < 16; off <<= 1) ts += __shfl_xor(ts, off);
            float c = __expf(m[r] - mn);
            l[r] = l[r] * c + ts;
            m[r] = mn;
#pragma unroll
            for (int t = 0; t < 4; t++) O[t][r] *= c;
            int prow = r + (lowhalf ? 0 : 8);
            spw[prow * 32 + qr]      = (half_t)p0;
            spw[prow * 32 + 16 + qr] = (half_t)p1;
        }
        // P (16x32) as A fragment; V slab rows jp*32..+31 as B fragments
        const half_t* pp = spw + qr * 32;
        v16h ap = mk16(*(const v8h*)(pp + aoff), *(const v8h*)(pp + 16 + aoff));
#pragma unroll
        for (int t = 0; t < 4; t++) {
            const half_t* pv = sVt + (t * 16 + qr) * WIN + jp * 32 + boff;
            v16h bv = mk16(*(const v8h*)pv, *(const v8h*)(pv + 16));
            O[t] = __builtin_amdgcn_wmma_f32_16x16x32_f16(false, ap, false, bv, (short)0, O[t], false, false);
        }
    }

    // normalize and scatter back to (shift-adjusted) token order
#pragma unroll
    for (int r = 0; r < 8; r++) {
        int row = r + (lowhalf ? 0 : 8);
        int qi = wv * 16 + row;
        int tko = b * SEQ + ((w * WIN + qi + shift) & (SEQ - 1));
        float inv = 1.0f / l[r];
#pragma unroll
        for (int t = 0; t < 4; t++)
            ao[(size_t)tko * DIM + colBase + t * 16 + qr] = (half_t)(O[t][r] * inv);
    }
}

// ---------------------------------------------------------------- host
extern "C" void kernel_launch(void* const* d_in, const int* in_sizes, int n_in,
                              void* d_out, int out_size, void* d_ws, size_t ws_size,
                              hipStream_t stream) {
    const int*   x        = (const int*)d_in[0];
    const float* mr       = (const float*)d_in[1];
    const float* tok_emb  = (const float*)d_in[2];
    const float* pos_emb  = (const float*)d_in[3];
    const float* mask_emb = (const float*)d_in[4];
    const float* t_W      = (const float*)d_in[5];
    const float* t_b      = (const float*)d_in[6];
    const float* ln1_s    = (const float*)d_in[7];
    const float* ln1_b    = (const float*)d_in[8];
    const float* Wq = (const float*)d_in[9];   const float* bq = (const float*)d_in[10];
    const float* Wk = (const float*)d_in[11];  const float* bk = (const float*)d_in[12];
    const float* Wv = (const float*)d_in[13];  const float* bv = (const float*)d_in[14];
    const float* Wo = (const float*)d_in[15];  const float* bo = (const float*)d_in[16];
    const float* ln2_s = (const float*)d_in[17];
    const float* ln2_b = (const float*)d_in[18];
    const float* W1 = (const float*)d_in[19];  const float* b1 = (const float*)d_in[20];
    const float* W2 = (const float*)d_in[21];  const float* b2 = (const float*)d_in[22];
    const float* phase_embs = (const float*)d_in[23];
    const float* out_W = (const float*)d_in[24];
    const float* out_b = (const float*)d_in[25];
    float* out = (float*)d_out;

    char* ws = (char*)d_ws;
    size_t off = 0;
    auto alloc = [&](size_t bytes) -> char* {
        char* p = ws + off;
        off += (bytes + 255) & ~(size_t)255;
        return p;
    };
    const size_t TD = (size_t)T_TOK * DIM;
    float*  h     = (float*)alloc(TD * 4);
    float*  pin   = (float*)alloc(TD * 4);
    half_t* xn16  = (half_t*)alloc(TD * 2);
    half_t* q16   = (half_t*)alloc(TD * 2);
    half_t* k16   = (half_t*)alloc(TD * 2);
    half_t* v16   = (half_t*)alloc(TD * 2);
    half_t* ao16  = (half_t*)alloc(TD * 2);
    half_t* hid16 = (half_t*)alloc((size_t)T_TOK * DFF * 2);
    half_t* wqT   = (half_t*)alloc((size_t)DIM * DIM * 2);
    half_t* wkT   = (half_t*)alloc((size_t)DIM * DIM * 2);
    half_t* wvT   = (half_t*)alloc((size_t)DIM * DIM * 2);
    half_t* woT   = (half_t*)alloc((size_t)DIM * DIM * 2);
    half_t* w1T   = (half_t*)alloc((size_t)DIM * DFF * 2);
    half_t* w2T   = (half_t*)alloc((size_t)DFF * DIM * 2);
    half_t* owT   = (half_t*)alloc((size_t)DIM * VOCAB * 2);
    float*  tvec  = (float*)alloc((size_t)BATCH * DIM * 4);

    const int tpb = 256;
    const int edgeBlocks = (int)(TD / tpb);

    // weight packing (f32 -> f16 transposed)
    pack_weight_kernel<<<(DIM * DIM + tpb - 1) / tpb, tpb, 0, stream>>>(Wq, wqT, DIM, DIM);
    pack_weight_kernel<<<(DIM * DIM + tpb - 1) / tpb, tpb, 0, stream>>>(Wk, wkT, DIM, DIM);
    pack_weight_kernel<<<(DIM * DIM + tpb - 1) / tpb, tpb, 0, stream>>>(Wv, wvT, DIM, DIM);
    pack_weight_kernel<<<(DIM * DIM + tpb - 1) / tpb, tpb, 0, stream>>>(Wo, woT, DIM, DIM);
    pack_weight_kernel<<<(DIM * DFF + tpb - 1) / tpb, tpb, 0, stream>>>(W1, w1T, DIM, DFF);
    pack_weight_kernel<<<(DFF * DIM + tpb - 1) / tpb, tpb, 0, stream>>>(W2, w2T, DFF, DIM);
    pack_weight_kernel<<<(DIM * VOCAB + tpb - 1) / tpb, tpb, 0, stream>>>(out_W, owT, DIM, VOCAB);
    tvec_kernel<<<(BATCH * DIM + tpb - 1) / tpb, tpb, 0, stream>>>(mr, t_W, t_b, tvec);

    embed_kernel<<<edgeBlocks, tpb, 0, stream>>>(x, tok_emb, pos_emb, mask_emb, tvec, h);

    dim3 gD(T_TOK / 128, DIM / 128);     // N = 512
    dim3 gF(T_TOK / 128, DFF / 128);     // N = 2048
    dim3 gV(T_TOK / 128, VOCAB / 128);   // N = 256
    dim3 gAttn(NWIN, NH);

    for (int ph = 0; ph < PHASES; ph++) {
        int shift = (ph & 1) ? (WIN / 2) : 0;
        phase_begin_kernel<<<edgeBlocks, tpb, 0, stream>>>(h, pin, phase_embs + ph * DIM);
        ln_kernel<<<T_TOK / 8, tpb, 0, stream>>>(h, ln1_s, ln1_b, xn16);
        gemm_kernel<0><<<gD, tpb, 0, stream>>>(xn16, wqT, bq, q16, DIM, DIM);
        gemm_kernel<0><<<gD, tpb, 0, stream>>>(xn16, wkT, bk, k16, DIM, DIM);
        gemm_kernel<0><<<gD, tpb, 0, stream>>>(xn16, wvT, bv, v16, DIM, DIM);
        attn_kernel<<<gAttn, tpb, 0, stream>>>(q16, k16, v16, ao16, shift);
        gemm_kernel<2><<<gD, tpb, 0, stream>>>(ao16, woT, bo, h, DIM, DIM);      // h += attn@Wo+bo
        ln_kernel<<<T_TOK / 8, tpb, 0, stream>>>(h, ln2_s, ln2_b, xn16);
        gemm_kernel<1><<<gF, tpb, 0, stream>>>(xn16, w1T, b1, hid16, DIM, DFF);  // gelu
        gemm_kernel<2><<<gD, tpb, 0, stream>>>(hid16, w2T, b2, h, DFF, DIM);     // h += mlp
        phase_end_kernel<<<edgeBlocks, tpb, 0, stream>>>(h, pin);
    }

    tof16_kernel<<<edgeBlocks, tpb, 0, stream>>>(h, xn16);
    gemm_kernel<3><<<gV, tpb, 0, stream>>>(xn16, owT, out_b, out, DIM, VOCAB);
}
